// Features_50027779064163
// MI455X (gfx1250) — 8-bit matrix operand order repaired and checked by reading, compiled, not run on hardware
//
#include <hip/hip_runtime.h>
#include <hip/hip_bf16.h>
#include <stdint.h>

typedef __attribute__((ext_vector_type(8))) int   v8i;
typedef __attribute__((ext_vector_type(8))) float v8f;

#define NWIN_B  3600
#define FDIM    70
#define WSTRIDE 4352

__constant__ int c_dr[12] = {0,1,1,1, 0,1,2,1, 0,2,3,2};
__constant__ int c_dc[12] = {1,1,0,-1, 2,1,0,-1, 3,2,0,-2};

__device__ __forceinline__ float wsumf(float v){
  #pragma unroll
  for (int m=16;m>0;m>>=1) v += __shfl_xor(v,m,32);
  return v;
}
__device__ __forceinline__ float wmaxf(float v){
  #pragma unroll
  for (int m=16;m>0;m>>=1) v = fmaxf(v, __shfl_xor(v,m,32));
  return v;
}
__device__ __forceinline__ float wminf(float v){
  #pragma unroll
  for (int m=16;m>0;m>>=1) v = fminf(v, __shfl_xor(v,m,32));
  return v;
}
__device__ __forceinline__ uint32_t wsumu(uint32_t v){
  #pragma unroll
  for (int m=16;m>0;m>>=1) v += (uint32_t)__shfl_xor((int)v,m,32);
  return v;
}
__device__ __forceinline__ uint32_t onehot4(uint32_t q4, uint32_t mrep){
  uint32_t x = q4 ^ mrep;
  uint32_t t = ((x & 0x7F7F7F7Fu) + 0x7F7F7F7Fu) | x;
  return ((~t >> 7) & 0x01010101u) * 0x38u;
}
__device__ __forceinline__ void wfence(){ __builtin_amdgcn_wave_barrier(); }

__global__ __launch_bounds__(256) void feat_kernel(const float* __restrict__ x,
                                                   float* __restrict__ tot,
                                                   float* __restrict__ cntb){
  __shared__ __align__(16) unsigned char smem[8*WSTRIDE];
  const int lane = threadIdx.x & 31;
  const int wave = threadIdx.x >> 5;
  const int wid  = blockIdx.x * 8 + wave;
  unsigned char* Wb = smem + wave*WSTRIDE;
  unsigned char* qv = Wb;
  unsigned char* ia = Wb + 320;
  unsigned char* ja = Wb + 640;
  float* gmf = (float*)(Wb + 960);
  float* fw  = (float*)(Wb + 4032);

  const int b   = wid / NWIN_B;
  const int rem = wid - b*NWIN_B;
  const int wy  = rem / 60;
  const int wx  = rem - wy*60;
  const float* img = x + (size_t)(b*3 + 1)*65536 + (size_t)(wy*4)*256 + wx*4;

  float s=0.f, ssq=0.f, cf=0.f, mmax=-INFINITY, mmin=INFINITY;
  uint32_t hp0=0, hp1=0;
  #pragma unroll
  for (int t=0;t<10;++t){
    int idx = lane + 32*t;
    if (idx < 289){
      int r = idx/17, c = idx - r*17;
      float v = img[r*256 + c];
      v = (v < 10.f) ? 0.f : v;
      s += v; ssq += v*v;
      int q = 0;
      if (v != 0.f){
        cf += 1.f; mmax = fmaxf(mmax,v); mmin = fminf(mmin,v);
        q = 1 + (int)((v - 7.5f) * 0.025f);
        q = (q > 7) ? 7 : q;
        if (q <= 3)      hp0 += 1u << (10*(q-1));
        else if (q <= 6) hp1 += 1u << (10*(q-4));
      }
      qv[idx] = (unsigned char)q;
    }
  }
  s=wsumf(s); ssq=wsumf(ssq); cf=wsumf(cf);
  mmax=wmaxf(mmax); mmin=wminf(mmin);
  hp0=wsumu(hp0); hp1=wsumu(hp1);
  float rmax = (cf > 0.f) ? mmax : 0.f;
  float ns   = fmaxf(cf, 1.f);
  float mean = s/ns;
  float var  = fmaxf((ssq - 2.f*mean*s + cf*mean*mean)/ns, 0.f);
  float sd   = sqrtf(var);
  if (lane==0){
    fw[0]=mean; fw[1]=sd;
    fw[2]=(mmax-mean)/(sd+1e-9f);
    fw[3]=(mean-mmin)/(sd+1e-9f);
    fw[4]=(float)( hp0        & 0x3FF)*(1.f/289.f);
    fw[5]=(float)((hp0 >> 10) & 0x3FF)*(1.f/289.f);
    fw[6]=(float)((hp0 >> 20) & 0x3FF)*(1.f/289.f);
    fw[7]=(float)( hp1        & 0x3FF)*(1.f/289.f);
    fw[8]=(float)((hp1 >> 10) & 0x3FF)*(1.f/289.f);
    fw[9]=(float)((hp1 >> 20) & 0x3FF)*(1.f/289.f);
  }
  wfence();

  const int half = lane >> 4;
  const uint32_t mrep = (uint32_t)(lane & 15) * 0x01010101u;

  for (int o=0;o<12;++o){
    int dr = c_dr[o], dc = c_dc[o];
    int adc = dc < 0 ? -dc : dc;
    int Wd  = 17 - adc;
    int Np  = Wd * (17 - dr);
    int padN = (Np + 63) & ~63;

    wfence();
    for (int k=lane; k<padN; k+=32){
      unsigned char iv=255, jv=255;
      if (k < Np){
        int r = k / Wd, c = k - r*Wd;
        int ic = (dc<0) ? (c+adc) : c;
        int jc = (dc<0) ? c : (c+dc);
        iv = qv[r*17 + ic];
        jv = qv[(r+dr)*17 + jc];
      }
      ia[k]=iv; ja[k]=jv;
    }
    wfence();

    v8f acc = {};
    for (int ch=0; ch<padN; ch+=64){
      v8i Av, Bv;
      #pragma unroll
      for (int v=0;v<8;++v){
        int ka = ((v>>2)<<5) | (half<<4) | ((v&3)<<2);
        int kb = ((v>>2)<<5) | (half<<4) | ((v&3)<<2);
        uint32_t qa = *(const uint32_t*)(ia + ch + ka);
        uint32_t qb = *(const uint32_t*)(ja + ch + kb);
        Av[v] = (int)onehot4(qa, mrep);
        Bv[v] = (int)onehot4(qb, mrep);
      }
      acc = __builtin_amdgcn_wmma_f32_16x16x64_fp8_fp8(Av, Bv, (short)0, acc,
                                                       false, false);
    }
    {
      int n = lane & 15;
      float* go = gmf + o*64;
      if (half==0 && n<8){
        #pragma unroll
        for (int v=0;v<8;++v) go[v*8+n] = acc[v];
      }
    }
  }
  wfence();

  if (lane < 12){
    const float* C = gmf + lane*64;
    float sall = 0.f;
    const float4* C4 = (const float4*)C;
    #pragma unroll
    for (int c=0;c<16;++c){ float4 t4 = C4[c]; sall += t4.x+t4.y+t4.z+t4.w; }
    sall *= 2.f;
    float inv_all = 1.f / fmaxf(sall, 1e-12f);
    float s7=0.f, entn=0.f, scon=0.f, shom=0.f, sg2=0.f,
          smi=0.f, smii=0.f, smij=0.f;
    for (int i=1;i<8;++i){
      {
        float g = (C[i*8+i]*2.f) * inv_all;
        float I = (float)(i-1);
        s7 += g; entn += g*log2f(g+1e-8f);
        shom += g; sg2 += g*g;
        smi += g*I; smii += g*I*I; smij += g*I*I;
      }
      for (int j=i+1;j<8;++j){
        float P = C[i*8+j] + C[j*8+i];
        float g = P * inv_all;
        float d = (float)(i-j), dd = d*d;
        float I = (float)(i-1), J = (float)(j-1);
        s7   += 2.f*g;
        entn += 2.f*g*log2f(g+1e-8f);
        scon += 2.f*g*dd;
        shom += 2.f*g/(1.f+dd);
        sg2  += 2.f*g*g;
        smi  += g*(I+J);
        smii += g*(I*I+J*J);
        smij += 2.f*g*I*J;
      }
    }
    float inv7 = 1.f / fmaxf(s7, 1e-12f);
    float con    = scon*inv7;
    float hom    = shom*inv7;
    float energy = sqrtf(sg2)*inv7;
    float mi  = smi*inv7;
    float sdi = sqrtf(fmaxf(smii*inv7 - mi*mi, 0.f));
    float prod = sdi*sdi;
    float cov  = smij*inv7 - mi*mi;
    float corr = (prod < 1e-15f) ? 1.f : cov / fmaxf(prod, 1e-15f);
    int o = lane;
    fw[10+o]=con; fw[22+o]=hom; fw[34+o]=energy; fw[46+o]=corr; fw[58+o]=-entn;
  }
  wfence();

  if (rmax > 0.f){
    for (int f=lane; f<FDIM; f+=32){
      float v = fw[f];
      if (v != 0.f){
        atomicAdd(&tot [b*FDIM+f], v);
        atomicAdd(&cntb[b*FDIM+f], 1.f);
      }
    }
  }
}

__global__ void init_ws_k(float* ws){
  int i = blockIdx.x*blockDim.x + threadIdx.x;
  if (i < 2*8*FDIM) ws[i] = 0.f;
}

__global__ void final_k(const float* __restrict__ ws, float* __restrict__ out){
  int i = blockIdx.x*blockDim.x + threadIdx.x;
  if (i < 8*FDIM) out[i] = ws[i] / (ws[8*FDIM + i] + 1e-9f);
}

extern "C" void kernel_launch(void* const* d_in, const int* in_sizes, int n_in,
                              void* d_out, int out_size, void* d_ws, size_t ws_size,
                              hipStream_t stream) {
  (void)in_sizes; (void)n_in; (void)out_size; (void)ws_size;
  const float* x = (const float*)d_in[0];
  float* ws  = (float*)d_ws;
  float* tot = ws;
  float* cnt = ws + 8*FDIM;
  float* out = (float*)d_out;

  init_ws_k<<<dim3((2*8*FDIM + 255)/256), dim3(256), 0, stream>>>(ws);
  feat_kernel<<<dim3(8*NWIN_B/8), dim3(256), 0, stream>>>(x, tot, cnt);
  final_k<<<dim3((8*FDIM + 255)/256), dim3(256), 0, stream>>>(ws, out);
}
